// Attention_49211735278173
// MI455X (gfx1250) — matrix operand order repaired and checked by reading, not compiled, not run on hardware
//
#include <hip/hip_runtime.h>
#include <math.h>

typedef __attribute__((ext_vector_type(16))) __bf16 v16bf;
typedef __attribute__((ext_vector_type(8)))  __bf16 v8bf;
typedef __attribute__((ext_vector_type(8)))  float  v8f;

// ---------------- problem constants ----------------
constexpr int Bb   = 4;
constexpr int Nq   = 2304;   // H*W = 48*48
constexpr int Cc   = 512;
constexpr int NH   = 8;
constexpr int HD   = 64;
constexpr int NK   = 576;    // (48/2)*(48/2)
constexpr int WI   = 48;
constexpr float SCALE = 0.125f;  // hd^-0.5

// ---------------- workspace layout (bytes) ----------------
constexpr size_t SZ_QH    = (size_t)Bb*NH*Nq*128*2;      // bf16 [b,h,n,128]
constexpr size_t SZ_KCAT  = (size_t)Bb*NH*1152*128*2;    // bf16 [b,h,1152,128]  (transposed)
constexpr size_t SZ_VCAT  = (size_t)Bb*NH*128*NK*2;      // bf16 [b,h,128,576]   (transposed)
constexpr size_t SZ_ATT   = (size_t)Bb*NH*Nq*128*4;      // f32  [b,h,n,128]
constexpr size_t SZ_OCAT  = (size_t)Bb*Nq*1024*2;        // bf16 [9216,1024]
constexpr size_t SZ_SLOT1 = (size_t)Bb*Nq*1024*2;        // Xcat | Patches | KVf
constexpr size_t SZ_SLOT2 = (size_t)4096*1024*2;         // Bq | Bsr | Bkv | Bproj (transposed)
constexpr size_t SZ_SLOT3 = (size_t)Bb*Nq*1024*4;        // Qf | Yconv+Xk | Sf+P | Fout

constexpr size_t O_QH    = 0;
constexpr size_t O_KCAT  = O_QH   + SZ_QH;
constexpr size_t O_VCAT  = O_KCAT + SZ_KCAT;
constexpr size_t O_ATT   = O_VCAT + SZ_VCAT;
constexpr size_t O_OCAT  = O_ATT  + SZ_ATT;
constexpr size_t O_SLOT1 = O_OCAT + SZ_OCAT;
constexpr size_t O_SLOT2 = O_SLOT1 + SZ_SLOT1;
constexpr size_t O_SLOT3 = O_SLOT2 + SZ_SLOT2;
constexpr size_t O_BIAS  = O_SLOT3 + SZ_SLOT3;

// =====================================================================
// bf16 WMMA GEMM: C[f32, MxN] = A[bf16, MxK] * B[bf16, KxN] (+bias[N])
// B supplied TRANSPOSED: Bt[N,K] row-major (ldb = K-stride).
// Requires M%64==0, N%128==0, K%32==0, lda/ldb/ldc %8==0.
// 128 threads (4 waves); block tile 64x128; wave tile 32x64 = 2x4 WMMA.
// Double-buffered LDS: next k-tile prefetched to registers before the
// WMMA block, one barrier per k-step, global latency hidden under WMMA.
// =====================================================================
__device__ __forceinline__ v16bf cat8(v8bf lo, v8bf hi)
{
    return __builtin_shufflevector(lo, hi, 0,1,2,3,4,5,6,7,8,9,10,11,12,13,14,15);
}

template <bool HASBIAS>
__global__ __launch_bounds__(128)
void gemm_bf16_kernel(const __bf16* __restrict__ A, const __bf16* __restrict__ Bt,
                      float* __restrict__ C, const float* __restrict__ bias,
                      int K, int lda, int ldb, int ldc)
{
    __shared__ __bf16 As[2][64][40];    // 80B row stride, 16B-aligned sub-rows
    __shared__ __bf16 Bs[2][128][40];   // rows = output columns (B transposed)

    const int tid  = threadIdx.x;
    const int lane = tid & 31;
    const int wave = tid >> 5;
    const int hi   = lane >> 4;
    const int l15  = lane & 15;
    const int bm   = blockIdx.y * 64;
    const int bn   = blockIdx.x * 128;
    const int wm   = (wave >> 1) * 32;   // 0 or 32
    const int wn   = (wave & 1) * 64;    // 0 or 64

    v8f acc[2][4] = {};

    const int ar = tid >> 1;          // 0..63
    const int ac = (tid & 1) * 16;    // 0 or 16

    const __bf16* aptr = A  + (size_t)(bm + ar)  * lda + ac;
    const __bf16* bptr = Bt + (size_t)(bn + tid) * ldb;

    // ---- prologue: stage k-tile 0 into buffer 0 ----
    uint4 pa0, pa1, pb0, pb1, pb2, pb3;
    {
        const uint4* as = reinterpret_cast<const uint4*>(aptr);
        pa0 = as[0]; pa1 = as[1];
        const uint4* bs = reinterpret_cast<const uint4*>(bptr);
        pb0 = bs[0]; pb1 = bs[1]; pb2 = bs[2]; pb3 = bs[3];
        uint4* ad = reinterpret_cast<uint4*>(&As[0][ar][ac]);
        ad[0] = pa0; ad[1] = pa1;
        uint4* bd = reinterpret_cast<uint4*>(&Bs[0][tid][0]);
        bd[0] = pb0; bd[1] = pb1; bd[2] = pb2; bd[3] = pb3;
    }
    __syncthreads();

    const int nk = K >> 5;
    for (int kt = 0; kt < nk; ++kt) {
        const int cur = kt & 1;
        const bool more = (kt + 1) < nk;

        // prefetch next k-tile from global into registers (latency hidden by WMMA)
        if (more) {
            const uint4* as = reinterpret_cast<const uint4*>(aptr + (kt + 1) * 32);
            pa0 = as[0]; pa1 = as[1];
            const uint4* bs = reinterpret_cast<const uint4*>(bptr + (kt + 1) * 32);
            pb0 = bs[0]; pb1 = bs[1]; pb2 = bs[2]; pb3 = bs[3];
        }

        // A fragment: lanes 0-15 hold K {0..7,16..23}; lanes 16-31 hold K {8..15,24..31}
        v16bf a[2], b[4];
        const int ka = hi ? 8 : 0;
        #pragma unroll
        for (int mi = 0; mi < 2; ++mi) {
            const __bf16* p = &As[cur][wm + mi * 16 + l15][0];
            a[mi] = cat8(*reinterpret_cast<const v8bf*>(p + ka),
                         *reinterpret_cast<const v8bf*>(p + ka + 16));
        }
        // B fragment: lane = column n; lanes 0-15 hold K {0..7,16..23}; lanes 16-31 hold K {8..15,24..31}
        const int kb = hi ? 8 : 0;
        #pragma unroll
        for (int nj = 0; nj < 4; ++nj) {
            const __bf16* p = &Bs[cur][wn + nj * 16 + l15][kb];
            b[nj] = cat8(*reinterpret_cast<const v8bf*>(p),
                         *reinterpret_cast<const v8bf*>(p + 16));
        }

        #pragma unroll
        for (int mi = 0; mi < 2; ++mi)
            #pragma unroll
            for (int nj = 0; nj < 4; ++nj)
                acc[mi][nj] = __builtin_amdgcn_wmma_f32_16x16x32_bf16(
                    false, a[mi], false, b[nj], (short)0, acc[mi][nj], false, false);

        // stage prefetched tile into the other buffer; single barrier per step
        if (more) {
            uint4* ad = reinterpret_cast<uint4*>(&As[cur ^ 1][ar][ac]);
            ad[0] = pa0; ad[1] = pa1;
            uint4* bd = reinterpret_cast<uint4*>(&Bs[cur ^ 1][tid][0]);
            bd[0] = pb0; bd[1] = pb1; bd[2] = pb2; bd[3] = pb3;
            __syncthreads();
        }
    }

    // C/D layout: lane n = l15, VGPR r -> row r (lanes 0-15) or 8+r (lanes 16-31)
    const int rb = bm + wm + hi * 8;
    #pragma unroll
    for (int nj = 0; nj < 4; ++nj) {
        const int cb = bn + wn + nj * 16 + l15;
        const float bv = HASBIAS ? bias[cb] : 0.f;
        #pragma unroll
        for (int r = 0; r < 8; ++r) {
            C[(size_t)(rb + r)      * ldc + cb] = acc[0][nj][r] + bv;
            C[(size_t)(rb + 16 + r) * ldc + cb] = acc[1][nj][r] + bv;
        }
    }
}

// =====================================================================
// Pack / elementwise kernels
// =====================================================================

// [B*N, 2C] bf16 = [x_re | x_im]
__global__ void pack_xcat_kernel(const float* __restrict__ xr, const float* __restrict__ xi,
                                 __bf16* __restrict__ out)
{
    int idx = blockIdx.x * blockDim.x + threadIdx.x;
    const int total = Bb * Nq * 1024;
    if (idx >= total) return;
    int row = idx >> 10, col = idx & 1023;
    float v = (col < Cc) ? xr[(size_t)row * Cc + col] : xi[(size_t)row * Cc + (col - Cc)];
    out[idx] = (__bf16)v;
}

// complex W[Kc,Nc] (interleaved f32) -> TRANSPOSED real bf16 Bt[2Nc, 2Kc];
// underlying real B = [[Re,Im],[-Im,Re]], Bt[n][k] = B[k][n]
__global__ void pack_wmat_t_kernel(const float* __restrict__ Wc, __bf16* __restrict__ out,
                                   int Kc, int Nc)
{
    int idx = blockIdx.x * blockDim.x + threadIdx.x;
    int total = 4 * Kc * Nc;
    if (idx >= total) return;
    int row = idx / (2 * Kc);   // n index
    int col = idx % (2 * Kc);   // k index
    int hn = (row >= Nc), hk = (col >= Kc);
    int n = row - hn * Nc, k = col - hk * Kc;
    const float* p = Wc + 2 * ((size_t)k * Nc + n);
    float re = p[0], im = p[1];
    float v = hk ? (hn ? re : -im) : (hn ? im : re);
    out[idx] = (__bf16)v;
}

// sr_w complex [512][512][2][2] -> TRANSPOSED bf16 Bt[1024, 4096]; K index kk=c*4+p*2+q
__global__ void pack_bsr_t_kernel(const float* __restrict__ srw, __bf16* __restrict__ out)
{
    int idx = blockIdx.x * blockDim.x + threadIdx.x;
    const int total = 1024 * 4096;
    if (idx >= total) return;
    int row = idx >> 12, col = idx & 4095;     // row = n, col = k
    int hn = (row >= 512), hk = (col >= 2048);
    int d = row - hn * 512, kk = col - hk * 2048;
    int c = kk >> 2, p = (kk >> 1) & 1, q = kk & 1;
    const float* s = srw + 2 * ((((size_t)d * Cc + c) * 2 + p) * 2 + q);
    float re = s[0], im = s[1];
    float v = hk ? (hn ? re : -im) : (hn ? im : re);
    out[idx] = (__bf16)v;
}

// patch matrix [B*Nk, 4096] bf16 = [patch_re(2048) | patch_im(2048)]   (A operand)
__global__ void pack_patches_kernel(const float* __restrict__ xr, const float* __restrict__ xi,
                                    __bf16* __restrict__ out)
{
    int idx = blockIdx.x * blockDim.x + threadIdx.x;
    const int total = Bb * NK * 4096;
    if (idx >= total) return;
    int row = idx >> 12, col = idx & 4095;
    int b = row / NK, nk = row % NK;
    int ph = nk / 24, pw = nk % 24;
    int half = col >> 11, kk = col & 2047;
    int c = kk >> 2, p = (kk >> 1) & 1, q = kk & 1;
    int n = (ph * 2 + p) * WI + (pw * 2 + q);
    const float* src = half ? xi : xr;
    out[idx] = (__bf16)src[((size_t)b * Nq + n) * Cc + c];
}

// Qf f32 [9216, 1024]=[re|im] -> Qh bf16 [b,h,n,128]=[q_re(64)|q_im(64)]  (A operand)
__global__ void pack_qh_kernel(const float* __restrict__ Qf, __bf16* __restrict__ Qh)
{
    int idx = blockIdx.x * blockDim.x + threadIdx.x;
    const int total = Bb * NH * Nq * 128;
    if (idx >= total) return;
    int t = idx & 127;
    int n = (idx >> 7) % Nq;
    int h = (idx >> 7) / Nq % NH;
    int b = idx / (128 * Nq * NH);
    int scol = (t < HD) ? (h * HD + t) : (Cc + h * HD + (t - HD));
    Qh[idx] = (__bf16)Qf[((size_t)b * Nq + n) * 1024 + scol];
}

// complex LayerNorm over C (var = E[x^2]-mean^2, complex sqrt), fp32 math.
__global__ __launch_bounds__(256)
void layernorm_c_kernel(const float* __restrict__ Y, const float* __restrict__ gain,
                        const float* __restrict__ bias, __bf16* __restrict__ Xk)
{
    int row = blockIdx.x;
    const float* yre = Y + (size_t)row * 1024;
    const float* yim = yre + Cc;
    int tid = threadIdx.x;

    float a0 = yre[tid],        b0 = yim[tid];
    float a1 = yre[tid + 256],  b1 = yim[tid + 256];
    __shared__ float r0[256], r1[256], r2[256], r3[256];
    r0[tid] = a0 + a1;
    r1[tid] = b0 + b1;
    r2[tid] = (a0 * a0 - b0 * b0) + (a1 * a1 - b1 * b1);
    r3[tid] = 2.f * (a0 * b0 + a1 * b1);
    __syncthreads();
    for (int s = 128; s > 0; s >>= 1) {
        if (tid < s) {
            r0[tid] += r0[tid + s];
            r1[tid] += r1[tid + s];
            r2[tid] += r2[tid + s];
            r3[tid] += r3[tid + s];
        }
        __syncthreads();
    }
    const float inv_c = 1.f / (float)Cc;
    float mr = r0[0] * inv_c, mi = r1[0] * inv_c;
    float vr = r2[0] * inv_c - (mr * mr - mi * mi);
    float vi = r3[0] * inv_c - 2.f * mr * mi;
    float zr = vr + 1e-5f, zi = vi;
    float rad = sqrtf(zr * zr + zi * zi);
    float sre = sqrtf(fmaxf(0.f, (rad + zr) * 0.5f));
    float sim = copysignf(sqrtf(fmaxf(0.f, (rad - zr) * 0.5f)), zi);
    float den = fmaxf(rad, 1e-30f);
    float ir = sre / den, ii = -sim / den;

    for (int c = tid; c < Cc; c += 256) {
        float xr = yre[c] - mr, xi2 = yim[c] - mi;
        float nr = xr * ir - xi2 * ii;
        float ni = xr * ii + xi2 * ir;
        float gr = gain[2 * c], gi = gain[2 * c + 1];
        float br = bias[2 * c], bi = bias[2 * c + 1];
        Xk[(size_t)row * 1024 + c]      = (__bf16)(gr * nr - gi * ni + br);
        Xk[(size_t)row * 1024 + Cc + c] = (__bf16)(gr * ni + gi * nr + bi);
    }
}

// KVf f32 [B*Nk, 2048]=[kv_re(1024)|kv_im(1024)]; k chans 0..511, v chans 512..1023,
// channel = h*64+d.
// Kcat (TRANSPOSED B): [b,h,1152,128]; row t: [k_re | -k_im], row 576+t: [k_im | k_re]
__global__ void pack_kcat_t_kernel(const float* __restrict__ KVf, __bf16* __restrict__ Kcat)
{
    int idx = blockIdx.x * blockDim.x + threadIdx.x;
    const int total = Bb * NH * 1152 * 128;
    if (idx >= total) return;
    int r = idx & 127;                       // k index within 128
    int col = (idx >> 7) % 1152;             // output column (key)
    int h = (idx >> 7) / 1152 % NH;
    int b = idx / (128 * 1152 * NH);
    int t = col % NK;
    int cim = (col >= NK);
    int d = r & 63, rhi = r >> 6;
    const float* base = KVf + ((size_t)b * NK + t) * 2048;
    float kre = base[h * HD + d];
    float kim = base[1024 + h * HD + d];
    float v = cim ? (rhi ? kre : kim) : (rhi ? -kim : kre);
    Kcat[idx] = (__bf16)v;
}

// Vcat (TRANSPOSED B): [b,h,128,576]; Vt[u][t], u<64: v_re, u>=64: v_im
__global__ void pack_vcat_t_kernel(const float* __restrict__ KVf, __bf16* __restrict__ Vcat)
{
    int idx = blockIdx.x * blockDim.x + threadIdx.x;
    const int total = Bb * NH * 128 * NK;
    if (idx >= total) return;
    int t = idx % NK;
    int u = (idx / NK) & 127;
    int h = idx / (NK * 128) % NH;
    int b = idx / (NK * 128 * NH);
    int d = u & 63, hi = u >> 6;
    Vcat[idx] = (__bf16)KVf[((size_t)b * NK + t) * 2048 + (hi ? 1024 : 0) + Cc + h * HD + d];
}

// softmax(|score| * scale) over 576 keys; S f32 [2304,1152]=[re|im] -> P bf16 [2304,576]
__global__ __launch_bounds__(256)
void softmax_amp_kernel(const float* __restrict__ S, __bf16* __restrict__ P)
{
    int row = blockIdx.x;
    const float* s = S + (size_t)row * 1152;
    int tid = threadIdx.x;
    float amp[3];
    float m = -3.4e38f;
    #pragma unroll
    for (int i = 0; i < 3; ++i) {
        int t = tid + i * 256;
        if (t < NK) {
            float re = s[t], im = s[NK + t];
            float a = SCALE * sqrtf(re * re + im * im);
            amp[i] = a;
            m = fmaxf(m, a);
        } else amp[i] = -3.4e38f;
    }
    __shared__ float red[256];
    red[tid] = m;
    __syncthreads();
    for (int st = 128; st > 0; st >>= 1) {
        if (tid < st) red[tid] = fmaxf(red[tid], red[tid + st]);
        __syncthreads();
    }
    float rm = red[0];
    __syncthreads();
    float e[3];
    float lsum = 0.f;
    #pragma unroll
    for (int i = 0; i < 3; ++i) {
        int t = tid + i * 256;
        if (t < NK) { e[i] = __expf(amp[i] - rm); lsum += e[i]; } else e[i] = 0.f;
    }
    red[tid] = lsum;
    __syncthreads();
    for (int st = 128; st > 0; st >>= 1) {
        if (tid < st) red[tid] += red[tid + st];
        __syncthreads();
    }
    float inv = 1.f / red[0];
    #pragma unroll
    for (int i = 0; i < 3; ++i) {
        int t = tid + i * 256;
        if (t < NK) P[(size_t)row * NK + t] = (__bf16)(e[i] * inv);
    }
}

// Att f32 [b,h,n,128]=[re(64)|im(64)] -> Ocat bf16 [9216,1024]=[re(C)|im(C)], chan c=h*64+d
__global__ void pack_ocat_kernel(const float* __restrict__ Att, __bf16* __restrict__ Ocat)
{
    int idx = blockIdx.x * blockDim.x + threadIdx.x;
    const int total = Bb * Nq * 1024;
    if (idx >= total) return;
    int row = idx >> 10, col = idx & 1023;
    int him = (col >= Cc);
    int c = col - him * Cc;
    int h = c >> 6, d = c & 63;
    int b = row / Nq, n = row % Nq;
    Ocat[idx] = (__bf16)Att[(((size_t)(b * NH + h)) * Nq + n) * 128 + him * HD + d];
}

// complex bias[C] (interleaved) -> f32 [2C] = [re | im]
__global__ void pack_cbias_kernel(const float* __restrict__ src, float* __restrict__ dst)
{
    int idx = blockIdx.x * blockDim.x + threadIdx.x;
    if (idx >= 1024) return;
    dst[idx] = (idx < Cc) ? src[2 * idx] : src[2 * (idx - Cc) + 1];
}

// Fout f32 [9216,1024]=[re|im] -> d_out interleaved complex64 view
__global__ void interleave_out_kernel(const float* __restrict__ Fout, float* __restrict__ out)
{
    int idx = blockIdx.x * blockDim.x + threadIdx.x;
    const int total = Bb * Nq * Cc;
    if (idx >= total) return;
    int r = idx / Cc, c = idx % Cc;
    out[2 * (size_t)idx]     = Fout[(size_t)r * 1024 + c];
    out[2 * (size_t)idx + 1] = Fout[(size_t)r * 1024 + Cc + c];
}

// =====================================================================
// host side
// =====================================================================
static inline int cdiv(int a, int b) { return (a + b - 1) / b; }

static void launch_gemm(const __bf16* A, const __bf16* Bt, float* C, const float* bias,
                        int M, int N, int K, int lda, int ldb, int ldc, hipStream_t s)
{
    dim3 grid(N / 128, M / 64);
    if (bias)
        gemm_bf16_kernel<true><<<grid, 128, 0, s>>>(A, Bt, C, bias, K, lda, ldb, ldc);
    else
        gemm_bf16_kernel<false><<<grid, 128, 0, s>>>(A, Bt, C, nullptr, K, lda, ldb, ldc);
}

extern "C" void kernel_launch(void* const* d_in, const int* in_sizes, int n_in,
                              void* d_out, int out_size, void* d_ws, size_t ws_size,
                              hipStream_t stream)
{
    const float* x_re  = (const float*)d_in[0];
    const float* x_im  = (const float*)d_in[1];
    const float* Wq    = (const float*)d_in[2];   // complex interleaved [512,512]
    const float* Wkv   = (const float*)d_in[3];   // [512,1024]
    const float* Wproj = (const float*)d_in[4];   // [512,512]
    const float* bproj = (const float*)d_in[5];   // [512]
    const float* srw   = (const float*)d_in[6];   // [512,512,2,2]
    const float* srb   = (const float*)d_in[7];   // [512]
    const float* gain  = (const float*)d_in[8];   // [512]
    const float* bias  = (const float*)d_in[9];   // [512]
    (void)in_sizes; (void)n_in; (void)out_size; (void)ws_size;

    char* ws = (char*)d_ws;
    __bf16* Qh     = (__bf16*)(ws + O_QH);
    __bf16* Kcat   = (__bf16*)(ws + O_KCAT);
    __bf16* Vcat   = (__bf16*)(ws + O_VCAT);
    float*  Att    = (float*) (ws + O_ATT);
    __bf16* Ocat   = (__bf16*)(ws + O_OCAT);
    // slot1: Xcat -> Patches -> KVf
    __bf16* Xcat    = (__bf16*)(ws + O_SLOT1);
    __bf16* Patches = (__bf16*)(ws + O_SLOT1);
    float*  KVf     = (float*) (ws + O_SLOT1);
    // slot2: weight matrices (transposed)
    __bf16* Bq    = (__bf16*)(ws + O_SLOT2);
    __bf16* Bsr   = (__bf16*)(ws + O_SLOT2);
    __bf16* Bkv   = (__bf16*)(ws + O_SLOT2);
    __bf16* Bproj = (__bf16*)(ws + O_SLOT2);
    // slot3: Qf -> (Yconv, Xk) -> (Sf, P) -> Fout
    float*  Qf    = (float*) (ws + O_SLOT3);
    float*  Yconv = (float*) (ws + O_SLOT3);
    __bf16* Xk    = (__bf16*)(ws + O_SLOT3 + (size_t)Nq * 1024 * 4);
    float*  Sf    = (float*) (ws + O_SLOT3);
    __bf16* P     = (__bf16*)(ws + O_SLOT3 + (size_t)Nq * 1152 * 4);
    float*  Fout  = (float*) (ws + O_SLOT3);
    float*  srbc  = (float*) (ws + O_BIAS);
    float*  bprc  = (float*) (ws + O_BIAS + 4096);

    const int TPB = 256;

    // ---- phase 1-2: q projection ----
    pack_xcat_kernel<<<cdiv(Bb * Nq * 1024, TPB), TPB, 0, stream>>>(x_re, x_im, Xcat);
    pack_wmat_t_kernel<<<cdiv(4 * Cc * Cc, TPB), TPB, 0, stream>>>(Wq, Bq, Cc, Cc);
    launch_gemm(Xcat, Bq, Qf, nullptr, Bb * Nq, 1024, 1024, 1024, 1024, 1024, stream);
    pack_qh_kernel<<<cdiv(Bb * NH * Nq * 128, TPB), TPB, 0, stream>>>(Qf, Qh);

    // ---- phase 3: sr conv as patch GEMM ----
    pack_patches_kernel<<<cdiv(Bb * NK * 4096, TPB), TPB, 0, stream>>>(x_re, x_im, Patches);
    pack_bsr_t_kernel<<<cdiv(1024 * 4096, TPB), TPB, 0, stream>>>(srw, Bsr);
    pack_cbias_kernel<<<4, TPB, 0, stream>>>(srb, srbc);
    launch_gemm(Patches, Bsr, Yconv, srbc, Bb * NK, 1024, 4096, 4096, 4096, 1024, stream);

    // ---- phase 4: complex layernorm ----
    layernorm_c_kernel<<<Bb * NK, 256, 0, stream>>>(Yconv, gain, bias, Xk);

    // ---- phase 5: kv projection ----
    pack_wmat_t_kernel<<<cdiv(4 * Cc * 1024, TPB), TPB, 0, stream>>>(Wkv, Bkv, Cc, 1024);
    launch_gemm(Xk, Bkv, KVf, nullptr, Bb * NK, 2048, 1024, 1024, 1024, 2048, stream);

    // ---- phase 6: per-head K/V packing (transposed B operands) ----
    pack_kcat_t_kernel<<<cdiv(Bb * NH * 1152 * 128, TPB), TPB, 0, stream>>>(KVf, Kcat);
    pack_vcat_t_kernel<<<cdiv(Bb * NH * 128 * NK, TPB), TPB, 0, stream>>>(KVf, Vcat);

    // ---- phase 7: attention per (b,h) ----
    for (int bh = 0; bh < Bb * NH; ++bh) {
        const __bf16* Qbh = Qh   + (size_t)bh * Nq * 128;
        const __bf16* Kbh = Kcat + (size_t)bh * 1152 * 128;
        const __bf16* Vbh = Vcat + (size_t)bh * 128 * NK;
        float*        Abh = Att  + (size_t)bh * Nq * 128;
        launch_gemm(Qbh, Kbh, Sf, nullptr, Nq, 1152, 128, 128, 128, 1152, stream);
        softmax_amp_kernel<<<Nq, 256, 0, stream>>>(Sf, P);
        launch_gemm(P, Vbh, Abh, nullptr, Nq, 128, NK, NK, NK, 128, stream);
    }

    // ---- phase 8-9: output projection ----
    pack_ocat_kernel<<<cdiv(Bb * Nq * 1024, TPB), TPB, 0, stream>>>(Att, Ocat);
    pack_wmat_t_kernel<<<cdiv(4 * Cc * Cc, TPB), TPB, 0, stream>>>(Wproj, Bproj, Cc, Cc);
    pack_cbias_kernel<<<4, TPB, 0, stream>>>(bproj, bprc);
    launch_gemm(Ocat, Bproj, Fout, bprc, Bb * Nq, 1024, 1024, 1024, 1024, 1024, stream);
    interleave_out_kernel<<<cdiv(Bb * Nq * Cc, TPB), TPB, 0, stream>>>(Fout, (float*)d_out);
}
